// GCNLayer_43860206027072
// MI455X (gfx1250) — matrix operand order repaired and checked by reading, not compiled, not run on hardware
//
#include <hip/hip_runtime.h>

typedef __bf16 v16bf __attribute__((ext_vector_type(16)));
typedef __bf16 v8bf  __attribute__((ext_vector_type(8)));
typedef float  v8f   __attribute__((ext_vector_type(8)));

#define NV 256      // N
#define DV 256      // D
#define CH 32       // j-rows per LDS chunk
#define ASTR 264    // LDS A-row stride in bf16 elems: 528 B = 132 dwords ≡ 4 (mod 64 banks)

__device__ __forceinline__ unsigned short f2bf(float f) {
    unsigned int u = __float_as_uint(f);
    u += 0x7FFFu + ((u >> 16) & 1u);          // round-to-nearest-even
    return (unsigned short)(u >> 16);
}
__device__ __forceinline__ float bf2f(unsigned short h) {
    return __uint_as_float(((unsigned int)h) << 16);
}

// ---------------- Kernel 0: repack w_alpha into WMMA B-fragment order (bf16 hi/lo) ----
// Fragment f = nt*8 + ks covers cols nt*16..+16, K ks*32..+32.
// Element (f, lane, e): n = nt*16 + (lane&15), k = ks*32 + ((lane>>4)<<4) + e.
// Stored at bt[f*512 + lane*16 + e] -> each lane's 16 bf16 are contiguous (32B), coalesced.
__global__ void repack_w(const float* __restrict__ w,
                         unsigned short* __restrict__ bthi,
                         unsigned short* __restrict__ btlo) {
    const int idx  = blockIdx.x * blockDim.x + threadIdx.x;   // 0..65535
    const int e    = idx & 15;
    const int lane = (idx >> 4) & 31;
    const int f    = idx >> 9;
    const int ks   = f & 7;
    const int nt   = f >> 3;
    const int n = nt * 16 + (lane & 15);
    const int k = ks * 32 + ((lane >> 4) << 4) + e;
    const float w0 = w[k * DV + n];
    const unsigned short hi = f2bf(w0);
    bthi[idx] = hi;
    btlo[idx] = f2bf(w0 - bf2f(hi));
}

// ---------------- Kernel 1: xi = x @ w_vi, xj = x @ w_vj ----------------
__global__ void proj_kernel(const float* __restrict__ x,
                            const float* __restrict__ w_vi,
                            const float* __restrict__ w_vj,
                            float* __restrict__ xi,
                            float* __restrict__ xj) {
    __shared__ float xs[DV];
    const int row = blockIdx.x;
    const int t   = threadIdx.x;
    xs[t] = x[(size_t)row * DV + t];
    __syncthreads();
    float s1 = 0.f, s2 = 0.f;
#pragma unroll 4
    for (int d = 0; d < DV; ++d) {
        const float xv = xs[d];
        s1 = fmaf(xv, w_vi[d * DV + t], s1);
        s2 = fmaf(xv, w_vj[d * DV + t], s2);
    }
    xi[(size_t)row * DV + t] = s1;
    xj[(size_t)row * DV + t] = s2;
}

// ---------------- Kernel 2: fused H build + adj-reduce + bf16x3 H @ w_alpha ----------
// grid = B*N (one block per (b,i)), block = 256 threads (8 waves)
__global__ void fused_kernel(const float* __restrict__ alpha,
                             const float* __restrict__ adj,
                             const unsigned short* __restrict__ bthi,
                             const unsigned short* __restrict__ btlo,
                             const float* __restrict__ bias_h,
                             const float* __restrict__ xi,
                             const float* __restrict__ xj,
                             float* __restrict__ ah_out,
                             float* __restrict__ new_alpha) {
    __shared__ __align__(16) unsigned short HhiL[CH * ASTR];   // 16,896 B
    __shared__ __align__(16) unsigned short HloL[CH * ASTR];   // 16,896 B

    const int bi   = blockIdx.x;          // b*N + i
    const int b    = bi >> 8;
    const int t    = threadIdx.x;
    const int lane = t & 31;
    const int wave = t >> 5;
    const int jsub = t >> 6;              // 0..3: row group in H phase
    const int c4   = (t & 63) << 2;       // 4-column base in H phase

    const float4 xi4       = *(const float4*)&xi[(size_t)bi * DV + c4];
    const float* adj_row   = adj + (size_t)bi * NV;
    const float* alpha_blk = alpha + (size_t)bi * NV * DV;
    float*       outA      = new_alpha + (size_t)bi * NV * DV;

    // WMMA fragment helpers (wave32 layouts, ISA 7.12.2)
    const int mrow  = lane & 15;
    const int hh    = lane >> 4;
    const int mbase = hh << 3;
    const int ncol  = lane & 15;

    float ahp0 = 0.f, ahp1 = 0.f, ahp2 = 0.f, ahp3 = 0.f;

    for (int j0 = 0; j0 < NV; j0 += CH) {
        __syncthreads();   // previous chunk's GEMM readers done before LDS overwrite

        // ---- build H chunk (f32), split into bf16 hi/lo in LDS, accumulate adj*H ----
        for (int jj = jsub; jj < CH; jj += 4) {
            const int j = j0 + jj;
            const float4 al  = *(const float4*)&alpha_blk[(size_t)j * DV + c4];
            const float4 xjv = *(const float4*)&xj[((size_t)b * NV + j) * DV + c4];
            const float4 bh  = *(const float4*)&bias_h[j * DV + c4];
            const float h0 = fmaxf(xi4.x + xjv.x + al.x + bh.x, 0.f);
            const float h1 = fmaxf(xi4.y + xjv.y + al.y + bh.y, 0.f);
            const float h2 = fmaxf(xi4.z + xjv.z + al.z + bh.z, 0.f);
            const float h3 = fmaxf(xi4.w + xjv.w + al.w + bh.w, 0.f);

            const unsigned short e0 = f2bf(h0), e1 = f2bf(h1), e2 = f2bf(h2), e3 = f2bf(h3);
            const unsigned short g0 = f2bf(h0 - bf2f(e0)), g1 = f2bf(h1 - bf2f(e1));
            const unsigned short g2 = f2bf(h2 - bf2f(e2)), g3 = f2bf(h3 - bf2f(e3));
            *(uint2*)&HhiL[jj * ASTR + c4] =
                make_uint2((unsigned)e0 | ((unsigned)e1 << 16), (unsigned)e2 | ((unsigned)e3 << 16));
            *(uint2*)&HloL[jj * ASTR + c4] =
                make_uint2((unsigned)g0 | ((unsigned)g1 << 16), (unsigned)g2 | ((unsigned)g3 << 16));

            const float aj = adj_row[j];
            ahp0 = fmaf(aj, h0, ahp0);
            ahp1 = fmaf(aj, h1, ahp1);
            ahp2 = fmaf(aj, h2, ahp2);
            ahp3 = fmaf(aj, h3, ahp3);
        }
        __syncthreads();

        // ---- bf16x3 GEMM: H_chunk (32 x 256) @ w_alpha (256 x 256) ----
        const v8f z = {0.f, 0.f, 0.f, 0.f, 0.f, 0.f, 0.f, 0.f};
        v8f acc[2][2];
        acc[0][0] = z; acc[0][1] = z; acc[1][0] = z; acc[1][1] = z;

        for (int ks = 0; ks < 8; ++ks) {
            const int fb0 = ((2 * wave + 0) * 8 + ks) * 512 + lane * 16;
            const int fb1 = ((2 * wave + 1) * 8 + ks) * 512 + lane * 16;
            const v16bf bh0 = *(const v16bf*)(bthi + fb0);
            const v16bf bl0 = *(const v16bf*)(btlo + fb0);
            const v16bf bh1 = *(const v16bf*)(bthi + fb1);
            const v16bf bl1 = *(const v16bf*)(btlo + fb1);
#pragma unroll
            for (int mt = 0; mt < 2; ++mt) {
                const int aoff = (mt * 16 + mrow) * ASTR + ks * 32 + 16 * hh;
                const v8bf a0 = *(const v8bf*)(HhiL + aoff);
                const v8bf a1 = *(const v8bf*)(HhiL + aoff + 8);
                const v8bf l0 = *(const v8bf*)(HloL + aoff);
                const v8bf l1 = *(const v8bf*)(HloL + aoff + 8);
                const v16bf ahi = __builtin_shufflevector(a0, a1,
                    0,1,2,3,4,5,6,7,8,9,10,11,12,13,14,15);
                const v16bf alo = __builtin_shufflevector(l0, l1,
                    0,1,2,3,4,5,6,7,8,9,10,11,12,13,14,15);

                acc[mt][0] = __builtin_amdgcn_wmma_f32_16x16x32_bf16(
                    false, ahi, false, bh0, (short)0, acc[mt][0], false, false);
                acc[mt][0] = __builtin_amdgcn_wmma_f32_16x16x32_bf16(
                    false, ahi, false, bl0, (short)0, acc[mt][0], false, false);
                acc[mt][0] = __builtin_amdgcn_wmma_f32_16x16x32_bf16(
                    false, alo, false, bh0, (short)0, acc[mt][0], false, false);

                acc[mt][1] = __builtin_amdgcn_wmma_f32_16x16x32_bf16(
                    false, ahi, false, bh1, (short)0, acc[mt][1], false, false);
                acc[mt][1] = __builtin_amdgcn_wmma_f32_16x16x32_bf16(
                    false, ahi, false, bl1, (short)0, acc[mt][1], false, false);
                acc[mt][1] = __builtin_amdgcn_wmma_f32_16x16x32_bf16(
                    false, alo, false, bh1, (short)0, acc[mt][1], false, false);
            }
        }

        // ---- relu + store output chunk (D layout: m = mbase+v, n = ncol) ----
#pragma unroll
        for (int mt = 0; mt < 2; ++mt) {
#pragma unroll
            for (int nt = 0; nt < 2; ++nt) {
                const int n = (2 * wave + nt) * 16 + ncol;
#pragma unroll
                for (int v = 0; v < 8; ++v) {
                    const int m = j0 + mt * 16 + mbase + v;
                    outA[(size_t)m * DV + n] = fmaxf(acc[mt][nt][v], 0.f);
                }
            }
        }
    }

    // ---- reduce adj-weighted partials across the 4 row groups via LDS ----
    __syncthreads();
    float* red = (float*)HhiL;             // reuse 4 KB of LDS
    *(float4*)&red[jsub * NV + c4] = make_float4(ahp0, ahp1, ahp2, ahp3);
    __syncthreads();
    const float s = red[t] + red[NV + t] + red[2 * NV + t] + red[3 * NV + t];
    ah_out[(size_t)bi * DV + t] = s;
}

// ---------------- Kernel 3: new_x = relu(AH @ w_node) ----------------
__global__ void node_kernel(const float* __restrict__ ah,
                            const float* __restrict__ w_node,
                            float* __restrict__ new_x) {
    __shared__ float as[DV];
    const int row = blockIdx.x;
    const int t   = threadIdx.x;
    as[t] = ah[(size_t)row * DV + t];
    __syncthreads();
    float s = 0.f;
#pragma unroll 4
    for (int d = 0; d < DV; ++d)
        s = fmaf(as[d], w_node[d * DV + t], s);
    new_x[(size_t)row * DV + t] = fmaxf(s, 0.f);
}

extern "C" void kernel_launch(void* const* d_in, const int* in_sizes, int n_in,
                              void* d_out, int out_size, void* d_ws, size_t ws_size,
                              hipStream_t stream) {
    (void)in_sizes; (void)n_in; (void)out_size; (void)ws_size;
    const float* x       = (const float*)d_in[0];
    const float* alpha   = (const float*)d_in[1];
    const float* adj     = (const float*)d_in[2];
    const float* w_alpha = (const float*)d_in[3];
    const float* w_vi    = (const float*)d_in[4];
    const float* w_vj    = (const float*)d_in[5];
    const float* bias_h  = (const float*)d_in[6];
    const float* w_node  = (const float*)d_in[7];

    const int B    = 2;
    const int rows = B * NV;                 // 512

    float* xi = (float*)d_ws;                // rows*DV f32
    float* xj = xi + (size_t)rows * DV;      // rows*DV f32
    float* ah = xj + (size_t)rows * DV;      // rows*DV f32
    unsigned short* bthi = (unsigned short*)(ah + (size_t)rows * DV);   // 65536 bf16
    unsigned short* btlo = bthi + 65536;                                 // 65536 bf16

    float* new_x     = (float*)d_out;                // B*N*D
    float* new_alpha = new_x + (size_t)rows * DV;    // B*N*N*D

    repack_w   <<<DV, DV, 0, stream>>>(w_alpha, bthi, btlo);
    proj_kernel<<<rows, DV, 0, stream>>>(x, w_vi, w_vj, xi, xj);
    fused_kernel<<<rows, DV, 0, stream>>>(alpha, adj, bthi, btlo, bias_h, xi, xj, ah, new_alpha);
    node_kernel<<<rows, DV, 0, stream>>>(ah, w_node, new_x);
}
